// VQVAE_vel_43026982372003
// MI455X (gfx1250) — matrix operand order repaired and checked by reading, not compiled, not run on hardware
//
#include <hip/hip_runtime.h>
#include <cstdint>

#define BSIZE 262144

typedef __attribute__((ext_vector_type(16))) __bf16 v16bf;
typedef __attribute__((ext_vector_type(8)))  float  v8f;
typedef __attribute__((ext_vector_type(4)))  unsigned int v4u;

union FragU { v4u q[2]; v16bf v; };

// Native f32 -> bf16 (RNE) conversion; lowers to v_cvt on gfx1250.
__device__ __forceinline__ unsigned short f2bf(float f) {
  __bf16 h = (__bf16)f;
  return __builtin_bit_cast(unsigned short, h);
}
__device__ __forceinline__ float bf2f(unsigned short h) {
  unsigned int u = ((unsigned int)h) << 16;
  return __builtin_bit_cast(float, u);
}
__device__ __forceinline__ float elu1(float x) {
  return x > 0.f ? x : (__expf(x) - 1.f);
}

// ---------------------------------------------------------------------------
// Tensor Data Mover: async 2-D tile DMA global->LDS (gfx1250).
// This toolchain exposes the 6-arg builtin:
//   (uint32x4 g0, int32x8 g1, int32x4, int32x4, int32x8, i32 cpol)
// ---------------------------------------------------------------------------
#if defined(__has_builtin)
#if __has_builtin(__builtin_amdgcn_tensor_load_to_lds) && \
    __has_builtin(__builtin_amdgcn_s_wait_tensorcnt)
#define HAS_TDM 1
#endif
#endif
#ifndef HAS_TDM
#define HAS_TDM 0
#endif

#if HAS_TDM
typedef __attribute__((ext_vector_type(4))) unsigned int u32x4;
typedef __attribute__((ext_vector_type(8))) int i32x8;
typedef __attribute__((ext_vector_type(4))) int i32x4;

__device__ __forceinline__ unsigned lds_addr_of(const void* p) {
  return (unsigned)(unsigned long long)(uintptr_t)p;
}

// D# for a 2-D tile: tile_w x tile_h elements, source row stride stride_elems,
// OOB columns (>= tensor_w) read as zero. dsz_log: 0=1B,1=2B,2=4B,3=8B.
__device__ __forceinline__ void tdm_load_2d(unsigned lds_off, const void* gptr,
                                            unsigned dsz_log,
                                            unsigned tile_w, unsigned tile_h,
                                            unsigned tensor_w,
                                            unsigned stride_elems) {
  unsigned long long ga = (unsigned long long)(uintptr_t)gptr;
  u32x4 g0;
  g0[0] = 1u;                                   // count=1 (valid), user mode
  g0[1] = lds_off;                              // lds_addr
  g0[2] = (unsigned)ga;                         // global_addr[31:0]
  g0[3] = ((unsigned)(ga >> 32) & 0x01FFFFFFu)  // global_addr[56:32]
          | (2u << 30);                         // type=2 ("image")
  i32x8 g1;
  g1[0] = (int)(dsz_log << 16);                 // wg_mask=0, data_size
  g1[1] = (int)((tensor_w & 0xFFFFu) << 16);    // tensor_dim0[15:0] @ bits 63:48
  g1[2] = (int)((tensor_w >> 16) |              // tensor_dim0[31:16]
                ((tile_h & 0xFFFFu) << 16));    // tensor_dim1[15:0]
  g1[3] = (int)(((tile_h >> 16) & 0xFFFFu) |    // tensor_dim1[31:16]
                ((tile_w & 0xFFFFu) << 16));    // tile_dim0
  g1[4] = (int)(tile_h & 0xFFFFu);              // tile_dim1 (tile_dim2=0)
  g1[5] = (int)stride_elems;                    // tensor_dim0_stride[31:0]
  g1[6] = 0;                                    // stride0[47:32]=0, stride1 lo=0
  g1[7] = 0;
  __builtin_amdgcn_tensor_load_to_lds(g0, g1, (i32x4)0, (i32x4)0, (i32x8)0, 0);
}
#endif

// A fragment (16x32 bf16) from row-major LDS: rows mrow..mrow+15, cols kbase..kbase+31.
// ISA A layout: lanes 0-15 -> M=lane, K {0..7,16..23}; lanes 16-31 -> M=lane-16, K {8..15,24..31}.
__device__ __forceinline__ v16bf load_a(const unsigned short* lds, int stride,
                                        int mrow, int kbase, int lane) {
  int r = mrow + (lane & 15);
  int k0 = kbase + ((lane >> 4) << 3);
  const unsigned short* p = lds + r * stride + k0;
  FragU f;
  f.q[0] = *(const v4u*)(p);
  f.q[1] = *(const v4u*)(p + 16);
  return f.v;
}

// B fragment (32x16 bf16) from row-major LDS W: lanes 0-15 -> N = nbase+lane, K {0..7,16..23}; lanes 16-31 -> N = nbase+lane-16, K {8..15,24..31}.
__device__ __forceinline__ v16bf load_b(const unsigned short* lds, int stride,
                                        int kbase, int nbase, int lane) {
  const unsigned short* p = lds + (kbase + ((lane >> 4) << 3)) * stride + nbase + (lane & 15);
  v16bf v;
#pragma unroll
  for (int i = 0; i < 8; ++i) {
    v[i] = __builtin_bit_cast(__bf16, p[i * stride]);
    v[i + 8] = __builtin_bit_cast(__bf16, p[(16 + i) * stride]);
  }
  return v;
}

__device__ __forceinline__ v8f wmma_bf16(v16bf a, v16bf b, v8f c) {
  return __builtin_amdgcn_wmma_f32_16x16x32_bf16(false, a, false, b, (short)0, c,
                                                 false, false);
}

// ---------------------------------------------------------------------------
// Fused GEMM (+ optional input BN+ELU, + optional BN-stats epilogue).
// 128 rows per workgroup, 8 waves, each wave computes a 16 x (NT*16) strip.
// Activation tile staged into LDS by the Tensor Data Mover (async), weights
// staged by VALU (need f32->bf16 conversion), BN+ELU applied in-place in LDS.
// ---------------------------------------------------------------------------
template<int KP, int NT, int KREAL, bool IN_F32, bool STATS>
__global__ __launch_bounds__(256) void gemm_bn_kernel(
    const void* __restrict__ inA, const float* __restrict__ W,
    const float* __restrict__ bias,
    const float* __restrict__ bnScale, const float* __restrict__ bnShift,
    float* __restrict__ statsSum, float* __restrict__ statsSq,
    unsigned short* __restrict__ outB, float* __restrict__ outF, int NREAL)
{
  constexpr int NP = NT * 16;
  constexpr int KC = IN_F32 ? 32 : ((KP > 64) ? 64 : KP);  // K-chunk in LDS
  constexpr int XS = IN_F32 ? 128 * KC : 1;                // f32 DMA staging
  __shared__ unsigned short abuf[128 * KC];
  __shared__ unsigned short wbuf[KC * NP];
  __shared__ float xstage[XS];
  __shared__ float sSum[NP];
  __shared__ float sSq[NP];

  const int tid = threadIdx.x;
  const int rowbase = blockIdx.x * 128;
  const int wv = tid >> 5, lane = tid & 31;
  const int mrow = wv * 16;

  if constexpr (STATS) {
    if (tid < NP) { sSum[tid] = 0.f; sSq[tid] = 0.f; }
  }

  v8f acc[NT];
  v8f zero = {0.f, 0.f, 0.f, 0.f, 0.f, 0.f, 0.f, 0.f};
#pragma unroll
  for (int nt = 0; nt < NT; ++nt) acc[nt] = zero;

  for (int k0 = 0; k0 < KP; k0 += KC) {
#if HAS_TDM
    // async DMA of the activation tile into LDS (wave 0 issues)
    if (wv == 0) {
      if constexpr (IN_F32) {
        tdm_load_2d(lds_addr_of(xstage),
                    (const float*)inA + (size_t)rowbase * KREAL + k0,
                    2, KC, 128, (unsigned)(KREAL - k0), (unsigned)KREAL);
      } else {
        tdm_load_2d(lds_addr_of(abuf),
                    (const unsigned short*)inA + (size_t)rowbase * KP + k0,
                    1, KC, 128, KC, (unsigned)KP);
      }
    }
#endif
    // stage weight chunk (f32 -> bf16, zero-padded to KP x NP) while DMA runs
    __builtin_prefetch(W, 0, 0);
    for (int e = tid; e < KC * NP; e += 256) {
      int kk = k0 + e / NP;
      int n = e - (e / NP) * NP;
      float w = (kk < KREAL && n < NREAL) ? W[kk * NREAL + n] : 0.f;
      wbuf[e] = f2bf(w);
    }
#if HAS_TDM
    if (wv == 0) __builtin_amdgcn_s_wait_tensorcnt(0);
    __syncthreads();
    if constexpr (IN_F32) {
      for (int e = tid; e < 128 * KC; e += 256) abuf[e] = f2bf(xstage[e]);
    } else {
      for (int e = tid; e < 128 * KC; e += 256) {
        int c = k0 + (e & (KC - 1));
        float v = bf2f(abuf[e]) * bnScale[c] + bnShift[c];
        abuf[e] = f2bf(elu1(v));
      }
    }
#else
    if constexpr (IN_F32) {
      const float* X = (const float*)inA;
      for (int e = tid; e < 128 * KC; e += 256) {
        int r = e / KC, cc = e - r * KC;
        int c = k0 + cc;
        float v = (c < KREAL) ? X[(size_t)(rowbase + r) * KREAL + c] : 0.f;
        abuf[e] = f2bf(v);
      }
    } else {
      const unsigned short* H = (const unsigned short*)inA;
      for (int e = tid; e < 128 * KC; e += 256) {
        int r = e / KC, cc = e - r * KC;
        int c = k0 + cc;
        float v = bf2f(H[(size_t)(rowbase + r) * KP + c]) * bnScale[c] + bnShift[c];
        abuf[e] = f2bf(elu1(v));
      }
    }
#endif
    __syncthreads();
#pragma unroll
    for (int kt = 0; kt < KC / 32; ++kt) {
      v16bf a = load_a(abuf, KC, mrow, kt * 32, lane);
#pragma unroll
      for (int nt = 0; nt < NT; ++nt) {
        v16bf b = load_b(wbuf, NP, kt * 32, nt * 16, lane);
        acc[nt] = wmma_bf16(a, b, acc[nt]);
      }
    }
    __syncthreads();
  }

  // epilogue: D layout -> lane holds one column, rows v (lanes 0-15) / v+8 (16-31)
#pragma unroll
  for (int nt = 0; nt < NT; ++nt) {
    int c = nt * 16 + (lane & 15);
    float bv = (c < NREAL) ? bias[c] : 0.f;
    float s = 0.f, s2 = 0.f;
#pragma unroll
    for (int v = 0; v < 8; ++v) {
      float val = acc[nt][v] + bv;
      int r = rowbase + mrow + ((lane < 16) ? v : v + 8);
      if constexpr (STATS) {
        if (c < NREAL) {
          outB[(size_t)r * NREAL + c] = f2bf(val);
          s += val; s2 += val * val;
        }
      } else {
        if (c < NREAL) outF[(size_t)r * NREAL + c] = val;
      }
    }
    if constexpr (STATS) {
      if (c < NREAL) { atomicAdd(&sSum[c], s); atomicAdd(&sSq[c], s2); }
    }
  }
  if constexpr (STATS) {
    __syncthreads();
    if (tid < NREAL) {
      atomicAdd(&statsSum[tid], sSum[tid]);
      atomicAdd(&statsSq[tid], sSq[tid]);
    }
  }
}

// ---------------------------------------------------------------------------
// BN stats -> per-channel affine (scale, shift)
// ---------------------------------------------------------------------------
__global__ void finalize_bn_kernel(const float* __restrict__ sum,
                                   const float* __restrict__ sq,
                                   const float* __restrict__ g,
                                   const float* __restrict__ b,
                                   float* __restrict__ scale,
                                   float* __restrict__ shift, int C) {
  int c = threadIdx.x;
  if (c < C) {
    const float invB = 1.f / (float)BSIZE;
    float m = sum[c] * invB;
    float v = sq[c] * invB - m * m;
    float sc = g[c] * rsqrtf(v + 1e-5f);
    scale[c] = sc;
    shift[c] = b[c] - m * sc;
  }
}

// ---------------------------------------------------------------------------
// Fused: BN2+ELU -> z head, vel head, VQ argmin (WMMA distances), gather,
// decoder layer 1 (+ stats for BN3). 128 rows / WG.
// ---------------------------------------------------------------------------
__global__ __launch_bounds__(256) void mid_kernel(
    const unsigned short* __restrict__ h2pre,
    const float* __restrict__ scale2, const float* __restrict__ shift2,
    const float* __restrict__ mu_w, const float* __restrict__ mu_b,
    const float* __restrict__ vel_w, const float* __restrict__ vel_b,
    const float* __restrict__ codebook,
    const float* __restrict__ dw1, const float* __restrict__ db1,
    float* __restrict__ z_out, float* __restrict__ vel_out,
    float* __restrict__ quant_out,
    unsigned short* __restrict__ d1_out,
    float* __restrict__ statsSum, float* __restrict__ statsSq)
{
  __shared__ unsigned short hbuf[128 * 64];
  __shared__ unsigned short pmu[64 * 16];
  __shared__ unsigned short pvel[64 * 16];
  __shared__ unsigned short pcb[32 * 256];   // codebook^T half-chunk (256 codes)
  __shared__ unsigned short pdw1[32 * 64];
  __shared__ unsigned short zbuf[128 * 32];  // K padded 16->32
  __shared__ unsigned short qbuf[128 * 32];
  __shared__ float cnorm[512];
  __shared__ int idxbuf[128];
  __shared__ float sSum[64];
  __shared__ float sSq[64];

  const int tid = threadIdx.x;
  const int rowbase = blockIdx.x * 128;
  const int wv = tid >> 5, lane = tid & 31;
  const int mrow = wv * 16;

  if (tid < 64) { sSum[tid] = 0.f; sSq[tid] = 0.f; }

#if HAS_TDM
  // async DMA raw h tile (128x64 bf16, contiguous) into hbuf
  if (wv == 0) {
    tdm_load_2d(lds_addr_of(hbuf), h2pre + (size_t)rowbase * 64,
                1, 64, 128, 64, 64);
  }
#endif
  // stage small weights while DMA runs
  for (int e = tid; e < 64 * 16; e += 256) {
    int k = e >> 4, n = e & 15;
    pmu[e]  = f2bf(mu_w[k * 16 + n]);
    pvel[e] = f2bf(n < 3 ? vel_w[k * 3 + n] : 0.f);
  }
  for (int e = tid; e < 32 * 64; e += 256) {
    int k = e >> 6, n = e & 63;
    pdw1[e] = f2bf(k < 16 ? dw1[k * 64 + n] : 0.f);
  }
  for (int e = tid; e < 512; e += 256) {
    float s = 0.f;
    for (int k = 0; k < 16; ++k) { float v = codebook[e * 16 + k]; s += v * v; }
    cnorm[e] = s;
  }
#if HAS_TDM
  if (wv == 0) __builtin_amdgcn_s_wait_tensorcnt(0);
  __syncthreads();
  // in-place BN2 + ELU on the DMA'd tile
  for (int e = tid; e < 128 * 64; e += 256) {
    int c = e & 63;
    float v = bf2f(hbuf[e]) * scale2[c] + shift2[c];
    hbuf[e] = f2bf(elu1(v));
  }
#else
  for (int e = tid; e < 128 * 64; e += 256) {
    int c = e & 63;
    float v = bf2f(h2pre[(size_t)rowbase * 64 + e]) * scale2[c] + shift2[c];
    hbuf[e] = f2bf(elu1(v));
  }
#endif
  __syncthreads();

  // ---- z head: [128x64] @ [64x16] ----
  {
    v8f acc = {0.f, 0.f, 0.f, 0.f, 0.f, 0.f, 0.f, 0.f};
#pragma unroll
    for (int kt = 0; kt < 2; ++kt) {
      v16bf a = load_a(hbuf, 64, mrow, kt * 32, lane);
      v16bf b = load_b(pmu, 16, kt * 32, 0, lane);
      acc = wmma_bf16(a, b, acc);
    }
    int c = lane & 15;
    float bv = mu_b[c];
#pragma unroll
    for (int v = 0; v < 8; ++v) {
      int rl = mrow + ((lane < 16) ? v : v + 8);
      float val = acc[v] + bv;
      z_out[(size_t)(rowbase + rl) * 16 + c] = val;
      zbuf[rl * 32 + c] = f2bf(val);
      zbuf[rl * 32 + 16 + c] = 0;   // zero K-pad
    }
  }
  // ---- vel head: [128x64] @ [64x3] ----
  {
    v8f acc = {0.f, 0.f, 0.f, 0.f, 0.f, 0.f, 0.f, 0.f};
#pragma unroll
    for (int kt = 0; kt < 2; ++kt) {
      v16bf a = load_a(hbuf, 64, mrow, kt * 32, lane);
      v16bf b = load_b(pvel, 16, kt * 32, 0, lane);
      acc = wmma_bf16(a, b, acc);
    }
    int c = lane & 15;
    if (c < 3) {
      float bv = vel_b[c];
#pragma unroll
      for (int v = 0; v < 8; ++v) {
        int rl = mrow + ((lane < 16) ? v : v + 8);
        vel_out[(size_t)(rowbase + rl) * 3 + c] = acc[v] + bv;
      }
    }
  }

  // ---- quantizer: argmin_j (||e_j||^2 - 2 z.e_j); ||z||^2 is row-constant ----
  float minv[8];
  int mini[8];
#pragma unroll
  for (int v = 0; v < 8; ++v) { minv[v] = 3.4e38f; mini[v] = 0; }

  for (int half = 0; half < 2; ++half) {
    __syncthreads();
    for (int e = tid; e < 32 * 256; e += 256) {   // stage codebook^T chunk
      int k = e >> 8, n = e & 255;
      int code = half * 256 + n;
      pcb[e] = f2bf(k < 16 ? codebook[code * 16 + k] : 0.f);
    }
    __syncthreads();
    v16bf a = load_a(zbuf, 32, mrow, 0, lane);
#pragma unroll
    for (int nt = 0; nt < 16; ++nt) {
      v16bf b = load_b(pcb, 256, 0, nt * 16, lane);
      v8f acc = {0.f, 0.f, 0.f, 0.f, 0.f, 0.f, 0.f, 0.f};
      acc = wmma_bf16(a, b, acc);
      int code = half * 256 + nt * 16 + (lane & 15);
      float cn = cnorm[code];
#pragma unroll
      for (int v = 0; v < 8; ++v) {
        float d = cn - 2.f * acc[v];
        if (d < minv[v]) { minv[v] = d; mini[v] = code; }
      }
    }
  }
  // reduce across the 16 lanes of each half-wave (same row per half)
#pragma unroll
  for (int v = 0; v < 8; ++v) {
#pragma unroll
    for (int off = 1; off < 16; off <<= 1) {
      float ov = __shfl_xor(minv[v], off, 32);
      int   oi = __shfl_xor(mini[v], off, 32);
      if (ov < minv[v] || (ov == minv[v] && oi < mini[v])) {
        minv[v] = ov; mini[v] = oi;
      }
    }
  }
  if ((lane & 15) == 0) {
#pragma unroll
    for (int v = 0; v < 8; ++v)
      idxbuf[mrow + ((lane < 16) ? v : v + 8)] = mini[v];
  }
  __syncthreads();

  // gather quantized codes
  for (int e = tid; e < 128 * 16; e += 256) {
    int r = e >> 4, c = e & 15;
    int code = idxbuf[r];
    float val = codebook[code * 16 + c];
    quant_out[(size_t)(rowbase + r) * 16 + c] = val;
    qbuf[r * 32 + c] = f2bf(val);
    qbuf[r * 32 + 16 + c] = 0;
  }
  __syncthreads();

  // ---- decoder layer 1: q @ dw1 + db1, with BN3 stats ----
  {
    v16bf a = load_a(qbuf, 32, mrow, 0, lane);
#pragma unroll
    for (int nt = 0; nt < 4; ++nt) {
      v16bf b = load_b(pdw1, 64, 0, nt * 16, lane);
      v8f acc = {0.f, 0.f, 0.f, 0.f, 0.f, 0.f, 0.f, 0.f};
      acc = wmma_bf16(a, b, acc);
      int c = nt * 16 + (lane & 15);
      float bv = db1[c];
      float s = 0.f, s2 = 0.f;
#pragma unroll
      for (int v = 0; v < 8; ++v) {
        float val = acc[v] + bv;
        int r = rowbase + mrow + ((lane < 16) ? v : v + 8);
        d1_out[(size_t)r * 64 + c] = f2bf(val);
        s += val; s2 += val * val;
      }
      atomicAdd(&sSum[c], s);
      atomicAdd(&sSq[c], s2);
    }
  }
  __syncthreads();
  if (tid < 64) {
    atomicAdd(&statsSum[tid], sSum[tid]);
    atomicAdd(&statsSq[tid], sSq[tid]);
  }
}

// ---------------------------------------------------------------------------
extern "C" void kernel_launch(void* const* d_in, const int* in_sizes, int n_in,
                              void* d_out, int out_size, void* d_ws, size_t ws_size,
                              hipStream_t stream) {
  (void)in_sizes; (void)n_in; (void)out_size; (void)ws_size;
  const float* x     = (const float*)d_in[0];
  const float* ew1   = (const float*)d_in[1];
  const float* eb1   = (const float*)d_in[2];
  const float* eg1   = (const float*)d_in[3];
  const float* ebt1  = (const float*)d_in[4];
  const float* ew2   = (const float*)d_in[5];
  const float* eb2   = (const float*)d_in[6];
  const float* eg2   = (const float*)d_in[7];
  const float* ebt2  = (const float*)d_in[8];
  const float* mu_w  = (const float*)d_in[9];
  const float* mu_b  = (const float*)d_in[10];
  const float* vel_w = (const float*)d_in[11];
  const float* vel_b = (const float*)d_in[12];
  const float* dw1   = (const float*)d_in[13];
  const float* db1   = (const float*)d_in[14];
  const float* dg1   = (const float*)d_in[15];
  const float* dbt1  = (const float*)d_in[16];
  const float* dw2   = (const float*)d_in[17];
  const float* db2   = (const float*)d_in[18];
  const float* dg2   = (const float*)d_in[19];
  const float* dbt2  = (const float*)d_in[20];
  const float* dw3   = (const float*)d_in[21];
  const float* db3   = (const float*)d_in[22];
  const float* cb    = (const float*)d_in[23];

  float* out    = (float*)d_out;
  float* x_hat  = out;
  float* quant  = out + (size_t)BSIZE * 45;
  float* zout   = out + (size_t)BSIZE * 61;
  float* velout = out + (size_t)BSIZE * 77;

  char* ws = (char*)d_ws;
  float* stats = (float*)ws;      // 768 floats of sums/sumsqs
  float* bnp   = stats + 768;     // 768 floats of scale/shift
  unsigned short* h1  = (unsigned short*)(ws + 8192);
  unsigned short* h2  = h1 + (size_t)BSIZE * 128;
  unsigned short* dd1 = h2 + (size_t)BSIZE * 64;
  unsigned short* dd2 = dd1 + (size_t)BSIZE * 64;

  float *sum1 = stats,       *sq1 = stats + 128;
  float *sum2 = stats + 256, *sq2 = stats + 320;
  float *sum3 = stats + 384, *sq3 = stats + 448;
  float *sum4 = stats + 512, *sq4 = stats + 640;
  float *sc1 = bnp,       *sh1 = bnp + 128;
  float *sc2 = bnp + 256, *sh2 = bnp + 320;
  float *sc3 = bnp + 384, *sh3 = bnp + 448;
  float *sc4 = bnp + 512, *sh4 = bnp + 640;

  (void)hipMemsetAsync(stats, 0, 768 * sizeof(float), stream);

  dim3 grid(BSIZE / 128), blk(256);
  // encoder L1: x[B,225] @ ew1[225,128] (+stats)
  gemm_bn_kernel<256, 8, 225, true,  true ><<<grid, blk, 0, stream>>>(
      x, ew1, eb1, nullptr, nullptr, sum1, sq1, h1, nullptr, 128);
  finalize_bn_kernel<<<1, 128, 0, stream>>>(sum1, sq1, eg1, ebt1, sc1, sh1, 128);
  // encoder L2: BN1+ELU(h1) @ ew2[128,64] (+stats)
  gemm_bn_kernel<128, 4, 128, false, true ><<<grid, blk, 0, stream>>>(
      h1, ew2, eb2, sc1, sh1, sum2, sq2, h2, nullptr, 64);
  finalize_bn_kernel<<<1, 64, 0, stream>>>(sum2, sq2, eg2, ebt2, sc2, sh2, 64);
  // heads + VQ + decoder L1
  mid_kernel<<<grid, blk, 0, stream>>>(h2, sc2, sh2, mu_w, mu_b, vel_w, vel_b,
                                       cb, dw1, db1, zout, velout, quant,
                                       dd1, sum3, sq3);
  finalize_bn_kernel<<<1, 64, 0, stream>>>(sum3, sq3, dg1, dbt1, sc3, sh3, 64);
  // decoder L2: BN3+ELU(d1) @ dw2[64,128] (+stats)
  gemm_bn_kernel<64,  8, 64,  false, true ><<<grid, blk, 0, stream>>>(
      dd1, dw2, db2, sc3, sh3, sum4, sq4, dd2, nullptr, 128);
  finalize_bn_kernel<<<1, 128, 0, stream>>>(sum4, sq4, dg2, dbt2, sc4, sh4, 128);
  // output: BN4+ELU(d2) @ dw3[128,45] -> x_hat (f32)
  gemm_bn_kernel<128, 3, 128, false, false><<<grid, blk, 0, stream>>>(
      dd2, dw3, db3, sc4, sh4, nullptr, nullptr, nullptr, x_hat, 45);
}
